// GATLayer_51677046505794
// MI455X (gfx1250) — matrix operand order repaired and checked by reading, not compiled, not run on hardware
//
#include <hip/hip_runtime.h>
#include <hip/hip_bf16.h>

// GAT layer, flash-style. e[i,j] = s_i + d_j is rank-1 pre-lrelu and lrelu is
// monotone, so rowmax m_i = lrelu(s_i + max_j d_j). Single pass, no NxN matrix.
// Main GEMM (attn @ Wh, M=16384 x N=128 x K=16384) on v_wmma_f32_16x16x32_f16,
// with B chunks double-buffered in LDS via global_load_async_to_lds_b128.

#define NROWS 16384
#define DIM   128
#define KC    128          // K-chunk staged in LDS (32KB of f16 B-operand data)
#define NKB   (KC / 32)    // 16x16x32 k-blocks per chunk
#define NC    (NROWS / KC) // number of K-chunks

typedef __attribute__((ext_vector_type(16))) _Float16 v16h;
typedef __attribute__((ext_vector_type(8)))  float    v8f;
typedef __attribute__((ext_vector_type(4)))  float    v4f;

__device__ __forceinline__ float lrelu(float x) { return fmaxf(x, 0.01f * x); }

// ---- CDNA5 async global->LDS copies (ASYNCcnt) ------------------------------
// LDS byte offset for VDST = low 32 bits of the generic pointer to __shared__.
__device__ __forceinline__ void async_b128(unsigned lds_off, const void* gaddr) {
  asm volatile("global_load_async_to_lds_b128 %0, %1, off"
               :: "v"(lds_off), "v"(gaddr) : "memory");
}
__device__ __forceinline__ void async_b32(unsigned lds_off, const void* gaddr) {
  asm volatile("global_load_async_to_lds_b32 %0, %1, off"
               :: "v"(lds_off), "v"(gaddr) : "memory");
}
__device__ __forceinline__ void wait_async0() {
#if __has_builtin(__builtin_amdgcn_s_wait_asynccnt)
  __builtin_amdgcn_s_wait_asynccnt(0);
#else
  asm volatile("s_wait_asynccnt 0x0" ::: "memory");
#endif
}

// ---------------------------------------------------------------------------
// Kernel 1: Wh = h @ W ; s = Wh@a_src ; d = Wh@a_dst ; WhB = f16(Wh) scattered
// into WMMA B-operand order: lane = (n&15) + (k&16), slot = k&15 per 32-K block.
// ---------------------------------------------------------------------------
__global__ __launch_bounds__(128) void wh_kernel(const float* __restrict__ h,
                                                 const float* __restrict__ Wm,
                                                 const float* __restrict__ a_src,
                                                 const float* __restrict__ a_dst,
                                                 float* __restrict__ s_out,
                                                 float* __restrict__ d_out,
                                                 _Float16* __restrict__ WhB) {
  const int t   = threadIdx.x;
  const int row = blockIdx.x * 8 + (t >> 4);
  const int c0  = (t & 15) * 8;

  float acc[8];
#pragma unroll
  for (int i = 0; i < 8; ++i) acc[i] = 0.f;

  const v4f* h4 = (const v4f*)(h + (size_t)row * DIM);
#pragma unroll 4
  for (int k4 = 0; k4 < DIM / 4; ++k4) {
    const v4f hv = h4[k4];
#pragma unroll
    for (int u = 0; u < 4; ++u) {
      const int k = k4 * 4 + u;
      const v4f* w4 = (const v4f*)(Wm + (size_t)k * DIM + c0);
      const v4f wa = w4[0];
      const v4f wb = w4[1];
      const float hk = hv[u];
      acc[0] += hk * wa[0]; acc[1] += hk * wa[1];
      acc[2] += hk * wa[2]; acc[3] += hk * wa[3];
      acc[4] += hk * wb[0]; acc[5] += hk * wb[1];
      acc[6] += hk * wb[2]; acc[7] += hk * wb[3];
    }
  }

  float sp = 0.f, dp = 0.f;
#pragma unroll
  for (int c = 0; c < 8; ++c) {
    sp += acc[c] * a_src[c0 + c];
    dp += acc[c] * a_dst[c0 + c];
  }
#pragma unroll
  for (int off = 8; off >= 1; off >>= 1) {
    sp += __shfl_xor(sp, off, 32);
    dp += __shfl_xor(dp, off, 32);
  }
  if ((t & 15) == 0) { s_out[row] = sp; d_out[row] = dp; }

  const int kb      = row >> 5;
  const int laneAdd = (row & 16);
  const int slot    = row & 15;
#pragma unroll
  for (int c = 0; c < 8; ++c) {
    const int n = c0 + c;
    const int ct = n >> 4, nin = n & 15;
    WhB[(size_t)(((kb * 8 + ct) * 32) + nin + laneAdd) * 16 + slot] = (_Float16)acc[c];
  }
}

// ---------------------------------------------------------------------------
// Kernel 2: dmax = max_j d[j]
// ---------------------------------------------------------------------------
__global__ __launch_bounds__(256) void dmax_kernel(const float* __restrict__ d,
                                                   float* __restrict__ dmax) {
  __shared__ float red[256];
  float m = -1e30f;
  for (int i = threadIdx.x; i < NROWS; i += 256) m = fmaxf(m, d[i]);
  red[threadIdx.x] = m;
  __syncthreads();
  for (int s = 128; s > 0; s >>= 1) {
    if (threadIdx.x < s) red[threadIdx.x] = fmaxf(red[threadIdx.x], red[threadIdx.x + s]);
    __syncthreads();
  }
  if (threadIdx.x == 0) *dmax = red[0];
}

// ---------------------------------------------------------------------------
// Kernel 3: out = elu( softmax(lrelu(s_i + d_j)) @ Wh )
// 128 threads (4 waves); each wave: 16 rows x 128 cols (8 acc tiles).
// Double-buffered LDS chunks filled with async global->LDS copies.
// P per element: 2 FMA + max + v_exp_f32 (shift and log2e folded into constants).
// ---------------------------------------------------------------------------
__global__ __launch_bounds__(128) void attn_kernel(const _Float16* __restrict__ WhB,
                                                   const float* __restrict__ s_in,
                                                   const float* __restrict__ d_in,
                                                   const float* __restrict__ dmaxp,
                                                   float* __restrict__ out) {
  __shared__ __align__(32) _Float16 Bs[2][KC * DIM];  // 2 x 32 KB
  __shared__ __align__(16) float    dsl[2][KC];       // 2 x 512 B

  const int t    = threadIdx.x;
  const int lane = t & 31;
  const int wave = t >> 5;
  const int rowBase = blockIdx.x * 64 + wave * 16;
  const int hi8  = (lane & 16) >> 1;  // 0 for lanes 0-15, 8 for lanes 16-31

  const float L2E  = 1.4426950408889634f;
  const float dmax = dmaxp[0];
  const float sv   = s_in[rowBase + (lane & 15)];
  const float mL   = lrelu(sv + dmax) * L2E;  // exact row max (in log2 domain)
  const float c1   = sv * L2E - mL;           // arg1 = fma(d, L2E, c1)
  const float c2   = sv * (0.01f * L2E) - mL; // arg2 = fma(d, .01*L2E, c2)

  v8f acc[8] = {};
  float dsum = 0.f;

  const char* gB = (const char*)WhB;          // chunk = KC*DIM*2 = 32768 bytes

  // ---- async stage of chunk `kc` into buffer `buf` ----
  auto stage = [&](int buf, int kc) {
    const char* src = gB + (size_t)kc * (KC * DIM * 2);
    const unsigned ldsB = (unsigned)(uintptr_t)&Bs[buf][0];
#pragma unroll
    for (int u = 0; u < 16; ++u)
      async_b128(ldsB + t * 16 + u * 2048, src + t * 16 + u * 2048);
    async_b32((unsigned)(uintptr_t)&dsl[buf][0] + t * 4,
              (const char*)(d_in + kc * KC) + t * 4);
  };

  stage(0, 0);
  wait_async0();
  __syncthreads();

  for (int kc = 0; kc < NC; ++kc) {
    const int cur = kc & 1;
    if (kc + 1 < NC) stage(cur ^ 1, kc + 1);  // overlap copy of next chunk

    const _Float16* Bc = &Bs[cur][0];
    const v4f*      dv = (const v4f*)&dsl[cur][0];

#pragma unroll
    for (int kb = 0; kb < NKB; ++kb) {
      const int koff = kb * 32 + 2 * hi8;  // multiple of 16
      const v4f q0 = dv[(koff >> 2) + 0];
      const v4f q1 = dv[(koff >> 2) + 1];
      const v4f q2 = dv[(koff >> 2) + 2];
      const v4f q3 = dv[(koff >> 2) + 3];
      float dl[16];
#pragma unroll
      for (int e = 0; e < 4; ++e) {
        dl[e] = q0[e]; dl[4 + e] = q1[e]; dl[8 + e] = q2[e]; dl[12 + e] = q3[e];
      }

      // A generation: P = exp2(max(d*L2E + c1, d*0.01*L2E + c2)) in A slot order
      v16h a;
      float psum = 0.f;
#pragma unroll
      for (int e = 0; e < 16; ++e) {
        const float a1 = __builtin_fmaf(dl[e], L2E, c1);
        const float a2 = __builtin_fmaf(dl[e], 0.01f * L2E, c2);
        const float p  = __builtin_amdgcn_exp2f(fmaxf(a1, a2));
        psum += p;
        a[e] = (_Float16)p;
      }
      dsum += psum;

#pragma unroll
      for (int ct = 0; ct < 8; ++ct) {
        const v16h b = *(const v16h*)&Bc[(size_t)((kb * 8 + ct) * 32 + lane) * 16];
        acc[ct] = __builtin_amdgcn_wmma_f32_16x16x32_f16(
            false, a, false, b, (short)0, acc[ct], false, false);
      }
    }

    wait_async0();    // our async copies into the other buffer are done
    __syncthreads();  // everyone finished reading `cur` and staging `cur^1`
  }

  // combine denominator halves: lanes L and L+16 both hold row (L&15)
  dsum += __shfl_xor(dsum, 16, 32);

  // C/D layout: VGPR v holds row (v + hi8), col = ct*16 + (lane&15)
#pragma unroll
  for (int ct = 0; ct < 8; ++ct) {
#pragma unroll
    for (int v = 0; v < 8; ++v) {
      const int roff = v + hi8;
      const float den = __shfl(dsum, roff, 32);
      const float val = acc[ct][v] / den;
      const float o = val > 0.f ? val : __expf(val) - 1.f;  // elu (alpha=1)
      out[(size_t)(rowBase + roff) * DIM + ct * 16 + (lane & 15)] = o;
    }
  }
}

// ---------------------------------------------------------------------------
// Launch. Workspace layout (needs ~4.33 MB):
//   [0, 64KB)        s  (f32, N)
//   [64KB, 128KB)    d  (f32, N)
//   [128KB, +4B)     dmax
//   [128KB+256, ..)  WhB (f16, N*128, B-operand permuted order) = 4 MB
// ---------------------------------------------------------------------------
extern "C" void kernel_launch(void* const* d_in, const int* in_sizes, int n_in,
                              void* d_out, int out_size, void* d_ws, size_t ws_size,
                              hipStream_t stream) {
  const float* h     = (const float*)d_in[0];
  const float* W     = (const float*)d_in[1];
  const float* a_src = (const float*)d_in[2];
  const float* a_dst = (const float*)d_in[3];
  float* out = (float*)d_out;

  char* ws = (char*)d_ws;
  float*    s_   = (float*)(ws);
  float*    dv   = (float*)(ws + 65536);
  float*    dmax = (float*)(ws + 131072);
  _Float16* WhB  = (_Float16*)(ws + 131072 + 256);

  wh_kernel<<<NROWS / 8, 128, 0, stream>>>(h, W, a_src, a_dst, s_, dv, WhB);
  dmax_kernel<<<1, 256, 0, stream>>>(dv, dmax);
  attn_kernel<<<NROWS / 64, 128, 0, stream>>>(WhB, s_, dv, dmax, out);
}
